// InterCamProxy_43989055045832
// MI455X (gfx1250) — matrix operand order repaired and checked by reading, not compiled, not run on hardware
//
#include <hip/hip_runtime.h>
#include <math.h>

// ---------------------------------------------------------------------------
// Problem constants (match reference)
// ---------------------------------------------------------------------------
#define BROWS   256        // batch
#define DDIM    2048       // feature dim
#define NPROXY  16384      // proxies
#define K_HARD  50
#define TEMP    0.07f
#define INV_TEMP (1.0f / TEMP)
#define EPS_NRM 1e-12f

typedef __attribute__((ext_vector_type(16))) __bf16 v16bf;
typedef __attribute__((ext_vector_type(8)))  float  v8f;
typedef __attribute__((__vector_size__(16))) int    v4i;

typedef __attribute__((address_space(1))) v4i* gptr_v4i;
typedef __attribute__((address_space(3))) v4i* lptr_v4i;

// ---------------------------------------------------------------------------
// CDNA5 async copy global->LDS (ASYNCcnt) with builtin / inline-asm fallback
// ---------------------------------------------------------------------------
__device__ __forceinline__ void async_copy_b128(const float* g, float* l) {
#if defined(__has_builtin) && __has_builtin(__builtin_amdgcn_global_load_async_to_lds_b128)
  __builtin_amdgcn_global_load_async_to_lds_b128(
      (gptr_v4i)g, (lptr_v4i)l, 0, 0);
#else
  unsigned lofs =
      (unsigned)(unsigned long long)(__attribute__((address_space(3))) void*)l;
  asm volatile("global_load_async_to_lds_b128 %0, %1, off"
               :: "v"(lofs), "v"(g) : "memory");
#endif
}

__device__ __forceinline__ void wait_async0() {
#if defined(__has_builtin) && __has_builtin(__builtin_amdgcn_s_wait_asynccnt)
  __builtin_amdgcn_s_wait_asynccnt(0);
#else
  asm volatile("s_wait_asynccnt 0" ::: "memory");
#endif
}

__device__ __forceinline__ v16bf cvt16_bf16(float4 b0, float4 b1, float4 b2,
                                            float4 b3) {
  v16bf r;
  r[0]  = (__bf16)b0.x;  r[1]  = (__bf16)b0.y;
  r[2]  = (__bf16)b0.z;  r[3]  = (__bf16)b0.w;
  r[4]  = (__bf16)b1.x;  r[5]  = (__bf16)b1.y;
  r[6]  = (__bf16)b1.z;  r[7]  = (__bf16)b1.w;
  r[8]  = (__bf16)b2.x;  r[9]  = (__bf16)b2.y;
  r[10] = (__bf16)b2.z;  r[11] = (__bf16)b2.w;
  r[12] = (__bf16)b3.x;  r[13] = (__bf16)b3.y;
  r[14] = (__bf16)b3.z;  r[15] = (__bf16)b3.w;
  return r;
}

// ---------------------------------------------------------------------------
// Stage 0: L2-normalize rows of inputs, emit bf16-packed x  [256 x 2048]
// ---------------------------------------------------------------------------
__global__ __launch_bounds__(256) void normalize_bf16_kernel(
    const float* __restrict__ inputs, unsigned* __restrict__ xbf_u32) {
  const int row = blockIdx.x;
  const int tid = threadIdx.x;
  __shared__ float red[256];

  const float* r = inputs + (size_t)row * DDIM;
  float ss = 0.f;
  for (int i = tid; i < DDIM; i += 256) {
    float v = r[i];
    ss += v * v;
  }
  red[tid] = ss;
  __syncthreads();
  for (int off = 128; off > 0; off >>= 1) {
    if (tid < off) red[tid] += red[tid + off];
    __syncthreads();
  }
  const float scale = 1.0f / fmaxf(sqrtf(red[0]), EPS_NRM);

  unsigned* orow = xbf_u32 + (size_t)row * (DDIM / 2);
  for (int i = tid; i < DDIM / 2; i += 256) {
    float a = r[2 * i + 0] * scale;
    float b = r[2 * i + 1] * scale;
    union { __bf16 h[2]; unsigned u; } pk;
    pk.h[0] = (__bf16)a;
    pk.h[1] = (__bf16)b;
    orow[i] = pk.u;
  }
}

// ---------------------------------------------------------------------------
// Stage 1 helpers
// ---------------------------------------------------------------------------
struct AFrags { uint4 a[4][2]; };

__device__ __forceinline__ void load_afrags(AFrags& f,
    const unsigned short* __restrict__ xbf, int m0, int col, int k, int kA) {
#pragma unroll
  for (int t = 0; t < 4; ++t) {
    const unsigned short* ap = xbf + (size_t)(m0 + t * 16 + col) * DDIM + k + kA;
    f.a[t][0] = *(const uint4*)(ap);
    f.a[t][1] = *(const uint4*)(ap + 16);
  }
}

__device__ __forceinline__ void wmma_block(v8f (&acc)[4][2], const AFrags& f,
    const float* buf, int ng, int col, int kB) {
  v16bf bfrag[2];
#pragma unroll
  for (int j = 0; j < 2; ++j) {
    const int c = ng * 32 + j * 16 + col;       // column within 64-slab
    const float4* bp = (const float4*)&buf[c * 32 + kB];
    bfrag[j] = cvt16_bf16(bp[0], bp[1], bp[4], bp[5]);
  }
#pragma unroll
  for (int t = 0; t < 4; ++t) {
    union { uint4 u[2]; v16bf v; } afr;
    afr.u[0] = f.a[t][0];
    afr.u[1] = f.a[t][1];
#pragma unroll
    for (int j = 0; j < 2; ++j) {
      acc[t][j] = __builtin_amdgcn_wmma_f32_16x16x32_bf16(
          false, afr.v, false, bfrag[j], (short)0, acc[t][j], false, false);
    }
  }
}

// ---------------------------------------------------------------------------
// Stage 1: sims[256,16384] = (x_bf16 @ proxy^T) * (1/TEMP)
//
// Block = 256 threads (8 waves) owns a 64-wide N-slab x all 256 M rows.
//   wave = (N-half ng in 0..1) x (M-group mg in 0..3)
//   each wave: 4 M-tiles x 2 N-tiles = 8 WMMA acc tiles (64 VGPRs).
// Proxy chunk (64 rows x 32 K, fp32, 8 KB) async-copied to LDS; the K-loop
// is unrolled x2 so the two LDS buffers and two A register sets ping-pong
// with no register copies (no WMMA WAR hazard NOPs).
// Proxy leaves HBM exactly once (grid covers each N column once).
// ---------------------------------------------------------------------------
__global__ __launch_bounds__(256, 1) void gemm_wmma_bf16_kernel(
    const unsigned short* __restrict__ xbf,   // bf16 bits [256][2048]
    const float* __restrict__ proxy,          // f32 [16384][2048]
    float* __restrict__ sims) {               // f32 [256][16384]
  __shared__ float pbuf[2][64 * 32];          // 2 x 8 KB proxy chunks

  const int tid  = threadIdx.x;
  const int lane = tid & 31;
  const int wave = tid >> 5;
  const int n0   = blockIdx.x * 64;

  const int col  = lane & 15;        // position within 16-wide tile
  const int half = lane >> 4;        // 0: lanes 0-15, 1: lanes 16-31
  const int kA   = half * 8;         // A frag K sub-offset (16-bit A 16x32)
  const int kB   = half * 8;         // B frag K sub-offset (16-bit B 32x16, same order as A)

  const int ng = wave >> 2;          // 0..1  -> N offset ng*32
  const int mg = wave & 3;           // 0..3  -> M offset mg*64
  const int m0 = mg * 64;

  // staging: 256 threads cover 64 rows x 32 floats; 2 x b128 per thread
  const int srow = tid >> 2;          // 0..63
  const int sseg = (tid & 3) * 8;     // 0,8,16,24
  const float* gsrc = proxy + (size_t)(n0 + srow) * DDIM + sseg;
  float* lds0 = &pbuf[0][srow * 32 + sseg];
  float* lds1 = &pbuf[1][srow * 32 + sseg];

  // ---- prologue: stage chunk 0 into buf0, preload A fragments for k=0
  async_copy_b128(gsrc + 0, lds0 + 0);
  async_copy_b128(gsrc + 4, lds0 + 4);

  AFrags a0, a1;
  load_afrags(a0, xbf, m0, col, 0, kA);

  v8f acc[4][2] = {};

  wait_async0();
  __syncthreads();

  for (int k0 = 0; k0 < DDIM; k0 += 64) {
    // ---- phase A: compute on buf0 (K = k0); stage buf1 (K = k0+32)
    {
      const int nk = k0 + 32;   // always < DDIM inside loop
      async_copy_b128(gsrc + nk + 0, lds1 + 0);
      async_copy_b128(gsrc + nk + 4, lds1 + 4);
      load_afrags(a1, xbf, m0, col, nk, kA);
    }
    wmma_block(acc, a0, pbuf[0], ng, col, kB);
    wait_async0();
    __syncthreads();

    // ---- phase B: compute on buf1 (K = k0+32); stage buf0 (K = k0+64)
    {
      const int nk = k0 + 64;
      if (nk < DDIM) {
        async_copy_b128(gsrc + nk + 0, lds0 + 0);
        async_copy_b128(gsrc + nk + 4, lds0 + 4);
        load_afrags(a0, xbf, m0, col, nk, kA);
      }
    }
    wmma_block(acc, a1, pbuf[1], ng, col, kB);
    wait_async0();
    __syncthreads();
  }

  // ---- epilogue: C/D layout (VGPR r -> M = r + 8*half) -> sims * 1/TEMP
#pragma unroll
  for (int t = 0; t < 4; ++t) {
#pragma unroll
    for (int j = 0; j < 2; ++j) {
#pragma unroll
      for (int r = 0; r < 8; ++r) {
        const int m = m0 + t * 16 + half * 8 + r;
        const int n = n0 + ng * 32 + j * 16 + col;
        sims[(size_t)m * NPROXY + n] = acc[t][j][r] * INV_TEMP;
      }
    }
  }
}

// ---------------------------------------------------------------------------
// Stage 2: per-row exact top-50 negatives + logsumexp + mean-positive.
// ---------------------------------------------------------------------------
__global__ __launch_bounds__(256) void row_loss_kernel(
    const float* __restrict__ sims,
    const int* __restrict__ targets, const int* __restrict__ cams,
    const int* __restrict__ pids,    const int* __restrict__ cids,
    float* __restrict__ loss_row) {
  extern __shared__ unsigned char smem_raw[];
  float*    vals = (float*)smem_raw;                       // 64 KB
  unsigned* keys = (unsigned*)(smem_raw + NPROXY * 4);     // 64 KB

  __shared__ float    redf[256];
  __shared__ unsigned s_cnt;
  __shared__ float    sh_posMax, sh_negMax, sh_posSum, sh_posCnt;
  __shared__ float    sh_negE, sh_posE;

  const int b   = blockIdx.x;
  const int tid = threadIdx.x;
  const int tgt = targets[b];
  const int cam = cams[b];
  const float* srow = sims + (size_t)b * NPROXY;

  float posMax = -__builtin_inff(), negMax = -__builtin_inff();
  float posSum = 0.f, posCnt = 0.f;
  for (int i = tid; i < NPROXY; i += 256) {
    float s  = srow[i];
    int  pid = pids[i], cid = cids[i];
    bool neg = (pid != tgt);
    bool pos = (pid == tgt) && (cid != cam);
    vals[i] = s;
    unsigned key = 0u;
    if (neg) {
      unsigned u = __float_as_uint(s);
      key = (u >> 31) ? ~u : (u | 0x80000000u);
    }
    keys[i] = key;
    if (pos) { posSum += s; posCnt += 1.f; posMax = fmaxf(posMax, s); }
    if (neg) { negMax = fmaxf(negMax, s); }
  }

  redf[tid] = posMax; __syncthreads();
  for (int off = 128; off > 0; off >>= 1) {
    if (tid < off) redf[tid] = fmaxf(redf[tid], redf[tid + off]);
    __syncthreads();
  }
  if (tid == 0) sh_posMax = redf[0];
  __syncthreads();

  redf[tid] = negMax; __syncthreads();
  for (int off = 128; off > 0; off >>= 1) {
    if (tid < off) redf[tid] = fmaxf(redf[tid], redf[tid + off]);
    __syncthreads();
  }
  if (tid == 0) sh_negMax = redf[0];
  __syncthreads();

  redf[tid] = posSum; __syncthreads();
  for (int off = 128; off > 0; off >>= 1) {
    if (tid < off) redf[tid] += redf[tid + off];
    __syncthreads();
  }
  if (tid == 0) sh_posSum = redf[0];
  __syncthreads();

  redf[tid] = posCnt; __syncthreads();
  for (int off = 128; off > 0; off >>= 1) {
    if (tid < off) redf[tid] += redf[tid + off];
    __syncthreads();
  }
  if (tid == 0) sh_posCnt = redf[0];
  __syncthreads();

  // 32-step radix binary search: largest thr with count(key >= thr) >= 50
  unsigned thr = 0u;
  for (int bit = 31; bit >= 0; --bit) {
    unsigned cand = thr | (1u << bit);
    if (tid == 0) s_cnt = 0u;
    __syncthreads();
    unsigned c = 0;
    for (int i = tid; i < NPROXY; i += 256) c += (keys[i] >= cand) ? 1u : 0u;
    atomicAdd(&s_cnt, c);
    __syncthreads();
    if (s_cnt >= K_HARD) thr = cand;
    __syncthreads();
  }

  if (tid == 0) s_cnt = 0u;
  __syncthreads();
  const float m = fmaxf(sh_posMax, sh_negMax);
  unsigned cg = 0;
  float negE = 0.f, posE = 0.f;
  for (int i = tid; i < NPROXY; i += 256) {
    unsigned k = keys[i];
    float    v = vals[i];
    if (k > thr) { cg++; negE += expf(v - m); }
    int pid = pids[i];
    if (pid == tgt && cids[i] != cam) posE += expf(v - m);
  }
  atomicAdd(&s_cnt, cg);
  redf[tid] = negE; __syncthreads();
  for (int off = 128; off > 0; off >>= 1) {
    if (tid < off) redf[tid] += redf[tid + off];
    __syncthreads();
  }
  if (tid == 0) sh_negE = redf[0];
  __syncthreads();

  redf[tid] = posE; __syncthreads();
  for (int off = 128; off > 0; off >>= 1) {
    if (tid < off) redf[tid] += redf[tid + off];
    __syncthreads();
  }
  if (tid == 0) sh_posE = redf[0];
  __syncthreads();

  if (tid == 0) {
    float loss = 0.f;
    if (sh_posCnt > 0.f) {
      float negTot = sh_negE;
      if (thr != 0u) {
        unsigned u = (thr & 0x80000000u) ? (thr & 0x7FFFFFFFu) : ~thr;
        float tv = __uint_as_float(u);
        negTot += (float)(K_HARD - (int)s_cnt) * expf(tv - m);
      }
      float logZ = m + logf(sh_posE + negTot);
      loss = logZ - sh_posSum / sh_posCnt;
    }
    loss_row[b] = loss;
  }
}

// ---------------------------------------------------------------------------
// Stage 3: reduce 256 row losses -> scalar  (loss.sum() / B)
// ---------------------------------------------------------------------------
__global__ __launch_bounds__(256) void final_reduce_kernel(
    const float* __restrict__ loss_row, float* __restrict__ out) {
  __shared__ float red[256];
  const int tid = threadIdx.x;
  red[tid] = loss_row[tid];
  __syncthreads();
  for (int off = 128; off > 0; off >>= 1) {
    if (tid < off) red[tid] += red[tid + off];
    __syncthreads();
  }
  if (tid == 0) out[0] = red[0] / (float)BROWS;
}

// ---------------------------------------------------------------------------
// Launch
// ---------------------------------------------------------------------------
extern "C" void kernel_launch(void* const* d_in, const int* in_sizes, int n_in,
                              void* d_out, int out_size, void* d_ws, size_t ws_size,
                              hipStream_t stream) {
  (void)in_sizes; (void)n_in; (void)out_size; (void)ws_size;

  const float* inputs  = (const float*)d_in[0];
  const float* proxy   = (const float*)d_in[1];
  const int*   targets = (const int*)d_in[2];
  const int*   cams    = (const int*)d_in[3];
  const int*   pids    = (const int*)d_in[4];
  const int*   cids    = (const int*)d_in[5];

  char* ws = (char*)d_ws;
  float*    sims     = (float*)ws;                                  // 16 MB
  size_t    off      = (size_t)BROWS * NPROXY * sizeof(float);
  unsigned* xbf_u32  = (unsigned*)(ws + off);                       // 1 MB
  off += (size_t)BROWS * DDIM * sizeof(unsigned short);
  float*    loss_row = (float*)(ws + off);                          // 1 KB

  normalize_bf16_kernel<<<BROWS, 256, 0, stream>>>(inputs, xbf_u32);

  gemm_wmma_bf16_kernel<<<NPROXY / 64, 256, 0, stream>>>(
      (const unsigned short*)xbf_u32, proxy, sims);

  size_t smem = (size_t)NPROXY * (sizeof(float) + sizeof(unsigned)); // 128 KB
  row_loss_kernel<<<BROWS, 256, smem, stream>>>(
      sims, targets, cams, pids, cids, loss_row);

  final_reduce_kernel<<<1, 256, 0, stream>>>(loss_row, (float*)d_out);
}
